// CfC_51153060496070
// MI455X (gfx1250) — matrix operand order repaired and checked by reading, not compiled, not run on hardware
//
#include <hip/hip_runtime.h>
#include <hip/hip_bf16.h>
#include <math.h>

// ---------------- problem constants ----------------
static constexpr int Bdim = 256;   // batch
static constexpr int Ldim = 1024;  // sequence length
static constexpr int Cdim = 128;   // input size
static constexpr int Udim = 256;   // units
static constexpr int BBU  = 256;   // backbone units
static constexpr int KBB  = Cdim + Udim;   // 384 : backbone K
static constexpr int NKF_BB = KBB / 32;    // 12 k-fragments (K=32 per wmma)
static constexpr int NKF_HD = BBU / 32;    // 8 k-fragments
static constexpr int NT   = BBU / 16;      // 16 column tiles
static constexpr int WGT  = 512;           // 16 waves per workgroup

// packed weight layout (ushort/bf16 elements) in d_ws:
//   backbone: [nt(16)][kf(12)][lane(32)][e(16)]              -> 98304 elems
//   head g  : 98304 + g*65536 + [nt(16)][kf(8)][lane(32)][e(16)]
static constexpr int PACK_BB_ELEMS = NT * NKF_BB * 512;        // 98304
static constexpr int PACK_HD_ELEMS = NT * NKF_HD * 512;        // 65536
static constexpr int PACK_TOTAL    = PACK_BB_ELEMS + 4 * PACK_HD_ELEMS; // 360448

typedef __attribute__((ext_vector_type(16))) __bf16       v16bf;
typedef __attribute__((ext_vector_type(8)))  float        v8f;
typedef __attribute__((ext_vector_type(4)))  unsigned int v4u;

// explicit global address space so the LICM-blocking opaque pointer still
// lowers to global_load_* (saddr form, LOADcnt only) instead of flat_load_*
typedef const __attribute__((address_space(1))) unsigned short* gcu16p;
typedef const __attribute__((address_space(1))) v4u*            gcv4up;

union BFrag {
  v16bf        bf;
  v4u          q[2];
  unsigned int u[8];
};

__device__ __forceinline__ unsigned short f2bf(float f) {
  unsigned int u = __float_as_uint(f);
  u += 0x7FFFu + ((u >> 16) & 1u);       // round-to-nearest-even
  return (unsigned short)(u >> 16);
}

__device__ __forceinline__ float fast_tanh(float x) {
#if __has_builtin(__builtin_amdgcn_tanhf)
  return __builtin_amdgcn_tanhf(x);      // v_tanh_f32 (gfx1250 TRANS op)
#else
  return tanhf(x);
#endif
}

__device__ __forceinline__ float fast_rcp(float x) {
#if __has_builtin(__builtin_amdgcn_rcpf)
  return __builtin_amdgcn_rcpf(x);       // v_rcp_f32
#else
  return 1.0f / x;
#endif
}

__device__ __forceinline__ void load_bfrag(BFrag& f, gcu16p blk, int lane) {
  gcv4up q = (gcv4up)blk;
  f.q[0] = q[lane * 2 + 0];
  f.q[1] = q[lane * 2 + 1];
}

// ---------------------------------------------------------------------------
// Pack f32 row-major weights into per-lane WMMA B-fragment order (bf16).
// B-fragment layout used here (contiguous k, 16-bit, 32x16): lane l -> column n = l&15,
// K-half = l>>4; element e (0..15) -> K = kf*32 + (l>>4)*16 + e.
// ---------------------------------------------------------------------------
__global__ void pack_weights_kernel(const float* __restrict__ Wbb,
                                    const float* __restrict__ Wff1,
                                    const float* __restrict__ Wff2,
                                    const float* __restrict__ Wta,
                                    const float* __restrict__ Wtb,
                                    unsigned short* __restrict__ packed) {
  int tid = blockIdx.x * blockDim.x + threadIdx.x;
  if (tid >= PACK_TOTAL) return;
  const float* W;
  int kf, nt, idx;
  if (tid < PACK_BB_ELEMS) {
    W = Wbb; idx = tid;
    int blk = idx >> 9;            // 512 elems per (nt,kf) block
    kf = blk % NKF_BB; nt = blk / NKF_BB;
  } else {
    int g = (tid - PACK_BB_ELEMS) >> 16;         // 65536 per head
    idx   = (tid - PACK_BB_ELEMS) & 65535;
    W = (g == 0) ? Wff1 : (g == 1) ? Wff2 : (g == 2) ? Wta : Wtb;
    int blk = idx >> 9;
    kf = blk & (NKF_HD - 1); nt = blk >> 3;
  }
  int lane = (idx >> 4) & 31;
  int e    = idx & 15;
  int k = kf * 32 + ((lane >> 4) << 4) + e;
  int n = nt * 16 + (lane & 15);
  packed[tid] = f2bf(W[k * 256 + n]);   // all W are (K,256) row-major
}

// ---------------------------------------------------------------------------
// Persistent recurrent scan: 16 workgroups, one 16-row batch tile each.
// 16 waves; wave w owns output column tile w for all GEMMs of the step.
// ---------------------------------------------------------------------------
__global__ __launch_bounds__(WGT)
void cfc_scan_kernel(const float* __restrict__ x,     // (B,L,C)
                     const float* __restrict__ h0,    // (B,U)
                     const float* __restrict__ bbb,
                     const float* __restrict__ bff1,
                     const float* __restrict__ bff2,
                     const float* __restrict__ bta,
                     const float* __restrict__ btb,
                     const unsigned short* __restrict__ packed,
                     float* __restrict__ out,         // (B,L,U)
                     float* __restrict__ outhN) {     // (B,U)
  __shared__ unsigned short xh[16 * KBB];   // A staging: rows m=0..15, k=0..383 ([x_t, h])
  __shared__ unsigned short zl[16 * BBU];   // backbone activation staging (bf16)

  const int t    = threadIdx.x;
  const int lane = t & 31;
  const int w    = t >> 5;        // wave id == column tile id (0..15)
  const int half = lane >> 4;     // K-half / M-half selector
  const int nlo  = lane & 15;
  const int bbase = blockIdx.x * 16;

  // ---- stage h0 into xh[:,128:384]  (each thread: 8 contiguous elems)
  {
    int i = t * 8;                 // 0..4095
    int m = i >> 8, u = i & 255;
    const v4u* hp = (const v4u*)&h0[(size_t)(bbase + m) * Udim + u];
    v4u a = hp[0], b = hp[1];
    unsigned int* dst = (unsigned int*)&xh[m * KBB + Cdim + u];
    dst[0] = f2bf(__uint_as_float(a[0])) | ((unsigned)f2bf(__uint_as_float(a[1])) << 16);
    dst[1] = f2bf(__uint_as_float(a[2])) | ((unsigned)f2bf(__uint_as_float(a[3])) << 16);
    dst[2] = f2bf(__uint_as_float(b[0])) | ((unsigned)f2bf(__uint_as_float(b[1])) << 16);
    dst[3] = f2bf(__uint_as_float(b[2])) | ((unsigned)f2bf(__uint_as_float(b[3])) << 16);
  }

  const float bias_bb = bbb[w * 16 + nlo];
  const float bias_h[4] = { bff1[w * 16 + nlo], bff2[w * 16 + nlo],
                            bta [w * 16 + nlo], btb [w * 16 + nlo] };

  // per-thread staging coords for x_t (4 consecutive floats each)
  const int sx_i = t * 4;                 // 0..2047
  const int sx_m = sx_i >> 7, sx_c = sx_i & 127;

  for (int l = 0; l < Ldim; ++l) {
    // make the packed-weight base opaque per iteration: blocks LICM from
    // hoisting all B-fragment loads out of the timestep loop (spill disaster);
    // kept in address_space(1) so codegen stays global_load (not flat_load)
    gcu16p pstep = (gcu16p)packed;
    asm volatile("" : "+s"(pstep));

    // ---- stage x_t (f32 -> bf16) into xh[:,0:128], non-temporal read
    {
      const v4u* xp = (const v4u*)&x[((size_t)(bbase + sx_m) * Ldim + l) * Cdim + sx_c];
      v4u xv = __builtin_nontemporal_load(xp);
      unsigned int* dst = (unsigned int*)&xh[sx_m * KBB + sx_c];
      dst[0] = f2bf(__uint_as_float(xv[0])) | ((unsigned)f2bf(__uint_as_float(xv[1])) << 16);
      dst[1] = f2bf(__uint_as_float(xv[2])) | ((unsigned)f2bf(__uint_as_float(xv[3])) << 16);
    }
    if (l + 1 < Ldim) {  // prefetch next timestep's x tile -> global_prefetch
      __builtin_prefetch(&x[((size_t)(bbase + w) * Ldim + (l + 1)) * Cdim + lane * 4], 0, 1);
    }
    __syncthreads();

    // ---- phase 1: backbone  z = lecun_tanh([x,h] @ Wbb + bbb), wave w -> ntile w
    {
      // A-fragments (contiguous k = 16*half + e, as packed B): m = nlo, pair v -> K offset
      BFrag afr[NKF_BB];
      #pragma unroll
      for (int kf = 0; kf < NKF_BB; ++kf) {
        #pragma unroll
        for (int v = 0; v < 8; ++v) {
          int koff = 2 * v + 16 * half;
          afr[kf].u[v] = *(const unsigned int*)&xh[nlo * KBB + kf * 32 + koff];
        }
      }
      v8f acc;
      #pragma unroll
      for (int r = 0; r < 8; ++r) acc[r] = bias_bb;

      gcu16p pb = pstep + (size_t)w * (NKF_BB * 512);
      // depth-4 software pipeline on the L2-resident B-fragment stream
      BFrag bq[4];
      #pragma unroll
      for (int p = 0; p < 4; ++p) load_bfrag(bq[p], pb + p * 512, lane);
      #pragma unroll
      for (int kf = 0; kf < NKF_BB; ++kf) {
        acc = __builtin_amdgcn_wmma_f32_16x16x32_bf16(
            false, afr[kf].bf, false, bq[kf & 3].bf, (short)0, acc, false, false);
        if (kf + 4 < NKF_BB) load_bfrag(bq[kf & 3], pb + (kf + 4) * 512, lane);
      }
      // lecun_tanh, store z tile (C layout: n = nlo, m = r + 8*half)
      #pragma unroll
      for (int r = 0; r < 8; ++r) {
        float zv = 1.7159f * fast_tanh(0.666f * acc[r]);
        int m = r + 8 * half;
        zl[m * BBU + w * 16 + nlo] = f2bf(zv);
      }
    }
    __syncthreads();

    // ---- phase 2: four heads on z, wave w -> ntile w for all heads
    {
      BFrag afr[NKF_HD];
      #pragma unroll
      for (int kf = 0; kf < NKF_HD; ++kf) {
        #pragma unroll
        for (int v = 0; v < 8; ++v) {
          int koff = 2 * v + 16 * half;
          afr[kf].u[v] = *(const unsigned int*)&zl[nlo * BBU + kf * 32 + koff];
        }
      }
      v8f acc[4];
      #pragma unroll
      for (int g = 0; g < 4; ++g) {
        #pragma unroll
        for (int r = 0; r < 8; ++r) acc[g][r] = bias_h[g];
        gcu16p ph = pstep + PACK_BB_ELEMS + g * PACK_HD_ELEMS + w * (NKF_HD * 512);
        BFrag bq[4];
        #pragma unroll
        for (int p = 0; p < 4; ++p) load_bfrag(bq[p], ph + p * 512, lane);
        #pragma unroll
        for (int kf = 0; kf < NKF_HD; ++kf) {
          acc[g] = __builtin_amdgcn_wmma_f32_16x16x32_bf16(
              false, afr[kf].bf, false, bq[kf & 3].bf, (short)0, acc[g], false, false);
          if (kf + 4 < NKF_HD) load_bfrag(bq[kf & 3], ph + (kf + 4) * 512, lane);
        }
      }
      // combine: h = tanh(ff1)*(1-t) + t*tanh(ff2), t = sigmoid(tb - ta)
      #pragma unroll
      for (int r = 0; r < 8; ++r) {
        float ff1 = fast_tanh(acc[0][r]);
        float ff2 = fast_tanh(acc[1][r]);
        float ti  = fast_rcp(1.0f + __expf(acc[2][r] - acc[3][r])); // sigmoid(tb-ta)
        float hnew = ff1 * (1.0f - ti) + ti * ff2;
        int m = r + 8 * half;
        int n = w * 16 + nlo;
        __builtin_nontemporal_store(hnew, &out[((size_t)(bbase + m) * Ldim + l) * Udim + n]);
        xh[m * KBB + Cdim + n] = f2bf(hnew);     // feed next step
        if (l == Ldim - 1)
          __builtin_nontemporal_store(hnew, &outhN[(size_t)(bbase + m) * Udim + n]);
      }
    }
    __syncthreads();
  }
}

// ---------------------------------------------------------------------------
extern "C" void kernel_launch(void* const* d_in, const int* in_sizes, int n_in,
                              void* d_out, int out_size, void* d_ws, size_t ws_size,
                              hipStream_t stream) {
  const float* x    = (const float*)d_in[0];
  const float* h0   = (const float*)d_in[1];
  const float* Wbb  = (const float*)d_in[2];
  const float* bbb  = (const float*)d_in[3];
  const float* Wff1 = (const float*)d_in[4];
  const float* bff1 = (const float*)d_in[5];
  const float* Wff2 = (const float*)d_in[6];
  const float* bff2 = (const float*)d_in[7];
  const float* Wta  = (const float*)d_in[8];
  const float* bta  = (const float*)d_in[9];
  const float* Wtb  = (const float*)d_in[10];
  const float* btb  = (const float*)d_in[11];

  unsigned short* packed = (unsigned short*)d_ws;   // 720,896 bytes used
  float* out   = (float*)d_out;
  float* outhN = out + (size_t)Bdim * Ldim * Udim;

  hipLaunchKernelGGL(pack_weights_kernel,
                     dim3((PACK_TOTAL + 255) / 256), dim3(256), 0, stream,
                     Wbb, Wff1, Wff2, Wta, Wtb, packed);

  hipLaunchKernelGGL(cfc_scan_kernel,
                     dim3(Bdim / 16), dim3(WGT), 0, stream,
                     x, h0, bbb, bff1, bff2, bta, btb, packed, out, outhN);
}
